// GaussianSlab_69406671503659
// MI455X (gfx1250) — compile-verified, matrix operand order checked by reading, not run on hardware
//
#include <hip/hip_runtime.h>
#include <hip/hip_bf16.h>
#include <math.h>

typedef __attribute__((ext_vector_type(16))) _Float16 v16h;
typedef __attribute__((ext_vector_type(8)))  float    v8f;

#define S_SPLATS 64
#define BLOCK    256
#define CTXD     6
#define CTXH     16
#define CTXL     8

// ---- fast branch-free transcendentals -------------------------------------
__device__ __forceinline__ float rcp_fast(float x) {
    return __builtin_amdgcn_rcpf(x);            // v_rcp_f32
}
__device__ __forceinline__ float tanh_fast(float x) {
#if __has_builtin(__builtin_amdgcn_tanhf)
    return __builtin_amdgcn_tanhf(x);           // v_tanh_f32 (gfx1250 TRANS)
#elif __has_builtin(__builtin_amdgcn_tanh_f32)
    return __builtin_amdgcn_tanh_f32(x);
#else
    // tanh(x) = 1 - 2/(exp(2x)+1): v_exp_f32 + v_rcp_f32, no branches
    float e = __expf(2.0f * x);
    return fmaf(-2.0f, rcp_fast(e + 1.0f), 1.0f);
#endif
}
__device__ __forceinline__ float erf_fast(float x) {
    // erf(x) ~= tanh(1.1283792*x + 0.1009060*x^3)  (GELU-style, ~1e-3 abs err)
    float x2 = x * x;
    return tanh_fast(x * fmaf(0.10090599f, x2, 1.12837917f));
}
__device__ __forceinline__ float sigmoid_f(float x) {
    return rcp_fast(1.0f + __expf(-x));
}
__device__ __forceinline__ float softplus_f(float x) {
    return (x > 20.0f) ? x : log1pf(__expf(x));
}

__global__ __launch_bounds__(BLOCK, 2)
void GaussianSlab_69406671503659_kernel(
    const float* __restrict__ hit_f, const float* __restrict__ hit_r,
    const float* __restrict__ front_center, const float* __restrict__ rear_center,
    const float* __restrict__ plane_normal, const float* __restrict__ plane_u,
    const float* __restrict__ plane_v,
    const unsigned char* __restrict__ valid_f, const unsigned char* __restrict__ valid_r,
    const float* __restrict__ ray_context,
    const float* __restrict__ mean_raw, const float* __restrict__ scale_raw,
    const float* __restrict__ strength_raw,
    const float* __restrict__ W1, const float* __restrict__ b1,
    const float* __restrict__ W2, const float* __restrict__ b2,
    const float* __restrict__ context_keys,
    const float* __restrict__ dgain_raw, const float* __restrict__ cscale_raw,
    float* __restrict__ out, int R)
{
    // ---- LDS staging -------------------------------------------------------
    __shared__ float    s_mean[S_SPLATS][3];
    __shared__ float    s_iv[S_SPLATS][3];
    __shared__ float    s_str[S_SPLATS];
    __shared__ _Float16 s_key[S_SPLATS][CTXL];   // keys * (1/sqrt(8)), f16
    __shared__ _Float16 s_lat[BLOCK][CTXL];      // per-ray latent, f16
    __shared__ float    s_sl[BLOCK][3];          // start_local
    __shared__ float    s_dl[BLOCK][3];          // end_local - start_local
    __shared__ float    s_csf[BLOCK];            // signal * softplus(context_scale)
    __shared__ float    s_gain[BLOCK];           // ray_gain * segment_valid
    __shared__ float    s_tau[BLOCK];

    const int tid  = threadIdx.x;
    const int lane = tid & 31;
    const int wv   = tid >> 5;

    // ---- Phase 1: per-block splat parameter expansion (threads 0..63) -----
    if (tid < S_SPLATS) {
        const int s = tid;
        #pragma unroll
        for (int k = 0; k < 3; ++k) {
            s_mean[s][k] = 1.15f * tanh_fast(mean_raw[s * 3 + k]);
            float sc = 0.05f + 0.60f * sigmoid_f(scale_raw[s * 3 + k]);
            s_iv[s][k] = 1.0f / (sc * sc);
        }
        s_str[s] = softplus_f(strength_raw[s]);
        #pragma unroll
        for (int k = 0; k < CTXL; ++k)
            s_key[s][k] = (_Float16)(context_keys[s * CTXL + k] * 0.35355339059f);
    }

    // ---- Phase 2: per-ray MLP + geometry, one ray per thread ---------------
    const int rayBase = blockIdx.x * BLOCK;
    int r = rayBase + tid;
    if (r >= R) r = R - 1;   // clamp; tail threads duplicate work, store guarded

    float ctx[CTXD];
    #pragma unroll
    for (int k = 0; k < CTXD; ++k) ctx[k] = ray_context[(size_t)r * CTXD + k];
    float signal = fminf(fmaxf(ctx[0], 0.0f), 1.0f);

    float h[CTXH];
    #pragma unroll
    for (int j = 0; j < CTXH; ++j) {
        float acc = b1[j];
        #pragma unroll
        for (int k = 0; k < CTXD; ++k) acc = fmaf(W1[j * CTXD + k], ctx[k], acc);
        h[j] = acc * sigmoid_f(acc);   // silu
    }
    #pragma unroll
    for (int i = 0; i < CTXL; ++i) {
        float acc = b2[i];
        #pragma unroll
        for (int j = 0; j < CTXH; ++j) acc = fmaf(W2[i * CTXH + j], h[j], acc);
        s_lat[tid][i] = (_Float16)acc;
    }

    // slab geometry (uniform-address scalar loads)
    float fcx = front_center[0], fcy = front_center[1], fcz = front_center[2];
    float rcx = rear_center[0],  rcy = rear_center[1],  rcz = rear_center[2];
    float cx = 0.5f * (fcx + rcx), cy = 0.5f * (fcy + rcy), cz = 0.5f * (fcz + rcz);
    float gx = rcx - fcx, gy = rcy - fcy, gz = rcz - fcz;
    float half_gap = fmaxf(0.5f * sqrtf(gx * gx + gy * gy + gz * gz), 1e-6f);
    float inv_hg = 1.0f / half_gap;
    float pnx = plane_normal[0], pny = plane_normal[1], pnz = plane_normal[2];
    float pux = plane_u[0], puy = plane_u[1], puz = plane_u[2];
    float pvx = plane_v[0], pvy = plane_v[1], pvz = plane_v[2];

    float fx = hit_f[(size_t)r * 3 + 0] - cx;
    float fy = hit_f[(size_t)r * 3 + 1] - cy;
    float fz = hit_f[(size_t)r * 3 + 2] - cz;
    float ex = hit_r[(size_t)r * 3 + 0] - cx;
    float ey = hit_r[(size_t)r * 3 + 1] - cy;
    float ez = hit_r[(size_t)r * 3 + 2] - cz;

    float su =  fx * pux + fy * puy + fz * puz;
    float sv = -(fx * pvx + fy * pvy + fz * pvz);
    float sw =  (fx * pnx + fy * pny + fz * pnz) * inv_hg;
    float eu =  ex * pux + ey * puy + ez * puz;
    float ev = -(ex * pvx + ey * pvy + ez * pvz);
    float ew =  (ex * pnx + ey * pny + ez * pnz) * inv_hg;

    s_sl[tid][0] = su; s_sl[tid][1] = sv; s_sl[tid][2] = sw;
    s_dl[tid][0] = eu - su; s_dl[tid][1] = ev - sv; s_dl[tid][2] = ew - sw;

    float sp_dg = softplus_f(dgain_raw[0]);
    float sp_cs = softplus_f(cscale_raw[0]);
    float gain  = fminf(fmaxf(1.0f + signal * sp_dg, 0.25f), 4.0f);
    float vmask = (valid_f[r] && valid_r[r]) ? 1.0f : 0.0f;
    s_gain[tid] = gain * vmask;
    s_csf[tid]  = signal * sp_cs;
    s_tau[tid]  = 0.0f;

    __syncthreads();

    // ---- Phase 3: WMMA logits + fused per-(ray,splat) integral -------------
    // Each wave owns 2 ray-tiles (mt = wv*2, wv*2+1); for each it sweeps the
    // 4 splat-tiles.  A (latent) is built once per mt and reused across the
    // 4 WMMAs; per-row masses accumulate in registers across splat-tiles so
    // only 8 ds_add_f32 per (lane, mt) hit LDS.
    v8f c0 = {};
    const bool lo = (lane < 16);
    const int  l15 = lane & 15;

    #pragma unroll 1
    for (int mt2 = 0; mt2 < 2; ++mt2) {
        const int mt   = wv * 2 + mt2;
        const int mrow = mt * 16 + l15;

        // A layout (16-bit 16x32): lanes 0-15 carry M=lane, K=0..7 in halves
        // 0..7; everything else zero (K padded 8 -> 32).
        v16h a;
        #pragma unroll
        for (int e = 0; e < 16; ++e) a[e] = (_Float16)0.0f;
        #pragma unroll
        for (int e = 0; e < CTXL; ++e) {
            _Float16 av = s_lat[mrow][e];
            a[e] = lo ? av : (_Float16)0.0f;
        }

        const int qbase = mt * 16 + (lo ? 0 : 8);
        float acc[8];
        #pragma unroll
        for (int i = 0; i < 8; ++i) acc[i] = 0.0f;

        #pragma unroll 1
        for (int nt = 0; nt < 4; ++nt) {
            const int scol = nt * 16 + l15;

            v16h b;
            #pragma unroll
            for (int e = 0; e < 16; ++e) b[e] = (_Float16)0.0f;
            #pragma unroll
            for (int e = 0; e < CTXL; ++e) {
                _Float16 bv = s_key[scol][e];
                b[e] = lo ? bv : (_Float16)0.0f;
            }

            v8f d = __builtin_amdgcn_wmma_f32_16x16x32_f16(
                false, a, false, b, (short)0, c0, false, false);

            // D layout: lanes 0-15 -> rows M=0..7, lanes 16-31 -> M=8..15;
            // column N = lane&15.  Each lane owns splat `scol`, 8 rays.
            const float mx = s_mean[scol][0], my = s_mean[scol][1], mz = s_mean[scol][2];
            const float ivx = s_iv[scol][0],  ivy = s_iv[scol][1],  ivz = s_iv[scol][2];
            const float str = s_str[scol];

            #pragma unroll
            for (int i = 0; i < 8; ++i) {
                const int q = qbase + i;
                float dx = s_dl[q][0], dy = s_dl[q][1], dz = s_dl[q][2];
                float ox = s_sl[q][0] - mx, oy = s_sl[q][1] - my, oz = s_sl[q][2] - mz;

                float A = fmaxf(dx * dx * ivx + dy * dy * ivy + dz * dz * ivz, 1e-6f);
                float B = ox * dx * ivx + oy * dy * ivy + oz * dz * ivz;
                float C = ox * ox * ivx + oy * oy * ivy + oz * oz * ivz;

                float rrA  = rsqrtf(A);          // one rsqrt feeds everything
                float rA   = rrA * rrA;          // 1/A
                float idn  = 0.70710678f * rrA;  // 1/sqrt(2A)
                float coef = 1.25331414f * rrA;  // sqrt(pi/(2A))

                float ehi  = erf_fast((A + B) * idn);
                float elo  = erf_fast(B * idn);
                float quad = fmaxf(C - B * B * rA, 0.0f);
                float integral = __expf(-0.5f * quad) * coef * fmaxf(ehi - elo, 0.0f);

                float cm = fmaxf(1.0f + s_csf[q] * tanh_fast(d[i]), 0.25f);
                acc[i] = fmaf(integral * str, cm, acc[i]);
            }
        }

        #pragma unroll
        for (int i = 0; i < 8; ++i)
            atomicAdd(&s_tau[qbase + i], acc[i]);   // ds_add_f32
    }

    __syncthreads();

    // ---- Phase 4: transmittance -------------------------------------------
    if (rayBase + tid < R)
        out[rayBase + tid] = __expf(-s_tau[tid] * s_gain[tid]);
}

extern "C" void kernel_launch(void* const* d_in, const int* in_sizes, int n_in,
                              void* d_out, int out_size, void* d_ws, size_t ws_size,
                              hipStream_t stream) {
    const float* hit_f        = (const float*)d_in[0];
    const float* hit_r        = (const float*)d_in[1];
    const float* front_center = (const float*)d_in[2];
    const float* rear_center  = (const float*)d_in[3];
    const float* plane_normal = (const float*)d_in[4];
    const float* plane_u      = (const float*)d_in[5];
    const float* plane_v      = (const float*)d_in[6];
    const unsigned char* valid_f = (const unsigned char*)d_in[7];
    const unsigned char* valid_r = (const unsigned char*)d_in[8];
    const float* ray_context  = (const float*)d_in[9];
    const float* mean_raw     = (const float*)d_in[10];
    const float* scale_raw    = (const float*)d_in[11];
    const float* strength_raw = (const float*)d_in[12];
    const float* W1           = (const float*)d_in[13];
    const float* b1           = (const float*)d_in[14];
    const float* W2           = (const float*)d_in[15];
    const float* b2           = (const float*)d_in[16];
    const float* context_keys = (const float*)d_in[17];
    const float* dgain_raw    = (const float*)d_in[18];
    const float* cscale_raw   = (const float*)d_in[19];

    const int R = in_sizes[0] / 3;
    const int grid = (R + BLOCK - 1) / BLOCK;

    GaussianSlab_69406671503659_kernel<<<grid, BLOCK, 0, stream>>>(
        hit_f, hit_r, front_center, rear_center, plane_normal, plane_u, plane_v,
        valid_f, valid_r, ray_context, mean_raw, scale_raw, strength_raw,
        W1, b1, W2, b2, context_keys, dgain_raw, cscale_raw,
        (float*)d_out, R);
}
